// ReliableAttention_6743098655128
// MI455X (gfx1250) — compile-verified, matrix operand order checked by reading, not run on hardware
//
#include <hip/hip_runtime.h>
#include <math.h>

// ---------------------------------------------------------------------------
// Problem constants (from the reference)
// ---------------------------------------------------------------------------
#define BB   2
#define NN   8192
#define CC   256
#define KL   40          // landmarks
#define INV_SCALE (1.0f/16.0f)   // 1/sqrt(256)

typedef __bf16 bf16_t;
typedef bf16_t v16bf __attribute__((ext_vector_type(16)));
typedef float  v8f   __attribute__((ext_vector_type(8)));

__device__ __forceinline__ bf16_t f2bf(float f) { return (bf16_t)f; }

// Load a 16-element bf16 WMMA fragment for one lane.
// ISA 16-bit A-matrix 16x32 layout (B is symmetric, column-per-lane):
//   elements 0..7  -> K = kbase + h*8 + (0..7)
//   elements 8..15 -> K = kbase + 16 + h*8 + (0..7)
__device__ __forceinline__ v16bf load_frag(const float* __restrict__ rowPtr,
                                           int kbase, int h) {
    const float4* p0 = reinterpret_cast<const float4*>(rowPtr + kbase + h * 8);
    const float4* p1 = reinterpret_cast<const float4*>(rowPtr + kbase + 16 + h * 8);
    float4 f0 = p0[0], f1 = p0[1];
    float4 f2 = p1[0], f3 = p1[1];
    v16bf r;
    r[0]  = f2bf(f0.x); r[1]  = f2bf(f0.y); r[2]  = f2bf(f0.z); r[3]  = f2bf(f0.w);
    r[4]  = f2bf(f1.x); r[5]  = f2bf(f1.y); r[6]  = f2bf(f1.z); r[7]  = f2bf(f1.w);
    r[8]  = f2bf(f2.x); r[9]  = f2bf(f2.y); r[10] = f2bf(f2.z); r[11] = f2bf(f2.w);
    r[12] = f2bf(f3.x); r[13] = f2bf(f3.y); r[14] = f2bf(f3.z); r[15] = f2bf(f3.w);
    return r;
}

// ---------------------------------------------------------------------------
// 1) Fused Q/K/V projection:  out = query @ W^T + b   (3 GEMMs, bf16 WMMA)
//    One wave computes a 16(M) x 64(N) strip: 4 accumulators share A-fragment.
//    wave tasks = 3 proj * 1024 rowTiles * 4 colStrips = 12288 -> 1536 blocks.
// ---------------------------------------------------------------------------
__global__ __launch_bounds__(256) void proj_qkv_kernel(
    const float* __restrict__ query,
    const float* __restrict__ Wq, const float* __restrict__ bq,
    const float* __restrict__ Wk, const float* __restrict__ bk,
    const float* __restrict__ Wv, const float* __restrict__ bv,
    float* __restrict__ qo, float* __restrict__ ko, float* __restrict__ vo)
{
    const int wid  = blockIdx.x * 8 + (threadIdx.x >> 5);
    const int lane = threadIdx.x & 31;
    const int r = lane & 15;
    const int h = lane >> 4;

    const int rowTiles = (BB * NN) / 16;            // 1024
    const int perProj  = rowTiles * 4;              // 4096
    const int proj     = wid / perProj;
    const int rem      = wid % perProj;
    const int rowTile  = rem >> 2;
    const int colStrip = rem & 3;

    const float* W; const float* bias; float* out;
    if (proj == 0)      { W = Wq; bias = bq; out = qo; }
    else if (proj == 1) { W = Wk; bias = bk; out = ko; }
    else                { W = Wv; bias = bv; out = vo; }

    const int rowBase = rowTile * 16;
    const int colBase = colStrip * 64;

    const float* Arow = query + (size_t)(rowBase + r) * CC;
    const float* Brow[4];
#pragma unroll
    for (int ns = 0; ns < 4; ++ns)
        Brow[ns] = W + (size_t)(colBase + ns * 16 + r) * CC;

    v8f acc[4];
#pragma unroll
    for (int ns = 0; ns < 4; ++ns) acc[ns] = (v8f)(0.0f);

#pragma unroll
    for (int kt = 0; kt < 8; ++kt) {
        const int kbase = kt * 32;
        if (kt < 7) __builtin_prefetch(Arow + kbase + 32, 0, 1);  // global_prefetch
        v16bf a = load_frag(Arow, kbase, h);
#pragma unroll
        for (int ns = 0; ns < 4; ++ns) {
            v16bf b = load_frag(Brow[ns], kbase, h);
            acc[ns] = __builtin_amdgcn_wmma_f32_16x16x32_bf16(
                false, a, false, b, (short)0, acc[ns], false, false);
        }
    }

    // C/D layout: lanes 0-15 -> M = e, lanes 16-31 -> M = 8 + e; N = lane&15.
#pragma unroll
    for (int ns = 0; ns < 4; ++ns) {
        const int col = colBase + ns * 16 + r;
        const float bv_ = bias[col];
#pragma unroll
        for (int e = 0; e < 8; ++e) {
            const int row = rowBase + h * 8 + e;
            out[(size_t)row * CC + col] = acc[ns][e] + bv_;
        }
    }
}

// ---------------------------------------------------------------------------
// 2a) qsum partials: qpart[b][chunk][c] = sum over 256 rows of q
//     grid = B*32, block = 256 (thread = c)
// ---------------------------------------------------------------------------
__global__ __launch_bounds__(256) void qsum_partial_kernel(
    const float* __restrict__ q, float* __restrict__ qpart)
{
    const int b = blockIdx.x >> 5;
    const int chunk = blockIdx.x & 31;
    const int c = threadIdx.x;
    const int base = b * NN + chunk * 256;
    float acc = 0.0f;
    for (int i = 0; i < 256; ++i)
        acc += q[(size_t)(base + i) * CC + c];
    qpart[(size_t)(b * 32 + chunk) * CC + c] = acc;
}

// 2b) qsum reduce: grid = B, block = 256
__global__ __launch_bounds__(256) void qsum_reduce_kernel(
    const float* __restrict__ qpart, float* __restrict__ qsum)
{
    const int b = blockIdx.x, c = threadIdx.x;
    float acc = 0.0f;
    for (int ch = 0; ch < 32; ++ch)
        acc += qpart[(size_t)(b * 32 + ch) * CC + c];
    qsum[b * CC + c] = acc;
}

// ---------------------------------------------------------------------------
// 3) score[b][m] = (qsum[b] . k[b][m]) / 16    (wave per row)
// ---------------------------------------------------------------------------
__global__ __launch_bounds__(256) void score_kernel(
    const float* __restrict__ k, const float* __restrict__ qsum,
    float* __restrict__ score)
{
    const int wid  = blockIdx.x * 8 + (threadIdx.x >> 5);
    const int lane = threadIdx.x & 31;
    const int b = wid / NN;
    const float* krow = k + (size_t)wid * CC;
    const float* qs = qsum + b * CC;
    float acc = 0.0f;
#pragma unroll
    for (int j = 0; j < CC / 32; ++j)
        acc += krow[lane + 32 * j] * qs[lane + 32 * j];
#pragma unroll
    for (int off = 16; off > 0; off >>= 1)
        acc += __shfl_xor(acc, off, 32);
    if (lane == 0) score[wid] = acc * INV_SCALE;
}

// ---------------------------------------------------------------------------
// 4) top-40 per batch (iterated argmax) + gather selected = k[top_idx]
//    grid = B, block = 256
// ---------------------------------------------------------------------------
__global__ __launch_bounds__(256) void topk_gather_kernel(
    const float* __restrict__ score, const float* __restrict__ k,
    float* __restrict__ scratch, int* __restrict__ top_idx,
    float* __restrict__ selected)
{
    __shared__ float smax[256];
    __shared__ int   sidx[256];
    const int b = blockIdx.x, tid = threadIdx.x;

    for (int i = tid; i < NN; i += 256) scratch[b * NN + i] = score[b * NN + i];
    __syncthreads();

    for (int iter = 0; iter < KL; ++iter) {
        float best = -3.0e38f; int bi = NN;
        for (int i = tid; i < NN; i += 256) {
            float vv = scratch[b * NN + i];
            if (vv > best || (vv == best && i < bi)) { best = vv; bi = i; }
        }
        smax[tid] = best; sidx[tid] = bi;
        __syncthreads();
        for (int s = 128; s > 0; s >>= 1) {
            if (tid < s) {
                if (smax[tid + s] > smax[tid] ||
                    (smax[tid + s] == smax[tid] && sidx[tid + s] < sidx[tid])) {
                    smax[tid] = smax[tid + s]; sidx[tid] = sidx[tid + s];
                }
            }
            __syncthreads();
        }
        const int idx = sidx[0];
        if (tid == 0) {
            top_idx[b * KL + iter] = idx;
            scratch[b * NN + idx] = -3.0e38f;
        }
        selected[(size_t)(b * KL + iter) * CC + tid] =
            k[((size_t)b * NN + idx) * CC + tid];
        __syncthreads();
    }
}

// ---------------------------------------------------------------------------
// 5) sim = softmax(X @ selected^T / 16)  for X in {q, k}; wave per row.
//    selected staged in LDS transposed with stride 41 (bank-conflict free).
// ---------------------------------------------------------------------------
__global__ __launch_bounds__(256) void logits_softmax_kernel(
    const float* __restrict__ X, const float* __restrict__ selected,
    float* __restrict__ sim)
{
    __shared__ float sel[CC * 41];            // sel[c*41 + l]
    const int blocksPerB = NN / 8;            // 1024
    const int b = blockIdx.x / blocksPerB;
    const int rowInB = (blockIdx.x % blocksPerB) * 8 + (threadIdx.x >> 5);
    const int lane = threadIdx.x & 31;

    for (int i = threadIdx.x; i < KL * CC; i += 256) {
        const int l = i / CC, c = i % CC;
        sel[c * 41 + l] = selected[(size_t)b * KL * CC + i];
    }
    __syncthreads();

    const size_t row = (size_t)b * NN + rowInB;
    const float* x = X + row * CC;
    const int l2 = 32 + (lane & 7);
    float d0 = 0.0f, d1 = 0.0f;
    for (int c = 0; c < CC; ++c) {
        const float xv = x[c];
        d0 += xv * sel[c * 41 + lane];
        d1 += xv * sel[c * 41 + l2];
    }
    const float v0 = d0 * INV_SCALE;
    const float v1 = d1 * INV_SCALE;

    float m = v0;
    if (lane < 8) m = fmaxf(m, v1);
#pragma unroll
    for (int off = 16; off > 0; off >>= 1)
        m = fmaxf(m, __shfl_xor(m, off, 32));

    const float e0 = expf(v0 - m);
    const float e1 = (lane < 8) ? expf(v1 - m) : 0.0f;
    float s = e0 + e1;
#pragma unroll
    for (int off = 16; off > 0; off >>= 1)
        s += __shfl_xor(s, off, 32);
    const float inv = 1.0f / s;

    sim[row * KL + lane] = e0 * inv;               // landmarks 0..31
    if (lane < 8) sim[row * KL + 32 + lane] = e1 * inv;  // landmarks 32..39
}

// ---------------------------------------------------------------------------
// 6a) t partials: t_part[b][chunk][l][c] = sum over 128 rows sim_km[m][l]*v[m][c]
//     grid = B*64, block = 256 (thread = c)
// ---------------------------------------------------------------------------
__global__ __launch_bounds__(256) void t_partial_kernel(
    const float* __restrict__ sim_km, const float* __restrict__ v,
    float* __restrict__ t_part)
{
    const int b = blockIdx.x / 64;
    const int chunk = blockIdx.x % 64;
    const int c = threadIdx.x;
    float acc[KL];
#pragma unroll
    for (int l = 0; l < KL; ++l) acc[l] = 0.0f;

    const int m0 = chunk * 128;
    for (int i = 0; i < 128; ++i) {
        const size_t row = (size_t)b * NN + m0 + i;
        const float vv = v[row * CC + c];
        const float* srow = sim_km + row * KL;
#pragma unroll
        for (int l = 0; l < KL; ++l) acc[l] += srow[l] * vv;
    }
    float* tp = t_part + (size_t)(b * 64 + chunk) * KL * CC;
#pragma unroll
    for (int l = 0; l < KL; ++l) tp[l * CC + c] = acc[l];
}

// 6b) t reduce over 64 chunks: grid = B*KL*CC/256 = 80, block = 256
__global__ __launch_bounds__(256) void t_reduce_kernel(
    const float* __restrict__ t_part, float* __restrict__ t)
{
    const int idx = blockIdx.x * 256 + threadIdx.x;   // < B*KL*CC
    const int b = idx / (KL * CC);
    const int r = idx % (KL * CC);
    float acc = 0.0f;
    for (int ch = 0; ch < 64; ++ch)
        acc += t_part[(size_t)(b * 64 + ch) * KL * CC + r];
    t[idx] = acc;
}

// ---------------------------------------------------------------------------
// 7) fold Wo: t2[b][l][c] = sum_j t[b][l][j] * Wo[c][j]
//    grid = B*KL = 80, block = 256 (thread = c)
// ---------------------------------------------------------------------------
__global__ __launch_bounds__(256) void t2_kernel(
    const float* __restrict__ t, const float* __restrict__ Wo,
    float* __restrict__ t2)
{
    const int b = blockIdx.x / KL;
    const int l = blockIdx.x % KL;
    const int c = threadIdx.x;
    const float* trow = t + (size_t)(b * KL + l) * CC;
    const float* wrow = Wo + (size_t)c * CC;
    float acc = 0.0f;
#pragma unroll 8
    for (int j = 0; j < CC; ++j) acc += trow[j] * wrow[j];
    t2[(size_t)(b * KL + l) * CC + c] = acc;
}

// ---------------------------------------------------------------------------
// 8) out = query + bo + sim_qn @ t2    (t2 staged in LDS)
//    grid = B * (N/32) = 512, block = 256 (thread = c), 32 rows/block
// ---------------------------------------------------------------------------
__global__ __launch_bounds__(256) void out_kernel(
    const float* __restrict__ query, const float* __restrict__ sim_qn,
    const float* __restrict__ t2, const float* __restrict__ bo,
    float* __restrict__ out)
{
    __shared__ float t2s[KL * CC];
    const int blocksPerB = NN / 32;           // 256
    const int b = blockIdx.x / blocksPerB;
    const int r0 = (blockIdx.x % blocksPerB) * 32;

    for (int i = threadIdx.x; i < KL * CC; i += 256)
        t2s[i] = t2[(size_t)b * KL * CC + i];
    __syncthreads();

    const int c = threadIdx.x;
    const float biasv = bo[c];
    for (int i = 0; i < 32; ++i) {
        const size_t row = (size_t)b * NN + r0 + i;
        const float* srow = sim_qn + row * KL;
        float acc = query[row * CC + c] + biasv;
#pragma unroll
        for (int l = 0; l < KL; ++l) acc += srow[l] * t2s[l * CC + c];
        out[row * CC + c] = acc;
    }
}

// ---------------------------------------------------------------------------
// launch
// ---------------------------------------------------------------------------
extern "C" void kernel_launch(void* const* d_in, const int* in_sizes, int n_in,
                              void* d_out, int out_size, void* d_ws, size_t ws_size,
                              hipStream_t stream)
{
    const float* query = (const float*)d_in[0];
    const float* Wq = (const float*)d_in[1]; const float* bq = (const float*)d_in[2];
    const float* Wk = (const float*)d_in[3]; const float* bk = (const float*)d_in[4];
    const float* Wv = (const float*)d_in[5]; const float* bv = (const float*)d_in[6];
    const float* Wo = (const float*)d_in[7]; const float* bo = (const float*)d_in[8];
    float* out = (float*)d_out;

    // workspace carving (floats)
    float* base = (float*)d_ws;
    size_t off = 0;
    float* q        = base + off; off += (size_t)BB * NN * CC;       // 4.19M
    float* kbuf     = base + off; off += (size_t)BB * NN * CC;
    float* vbuf     = base + off; off += (size_t)BB * NN * CC;
    float* qpart    = base + off; off += (size_t)BB * 32 * CC;
    float* qsum     = base + off; off += (size_t)BB * CC;
    float* score    = base + off; off += (size_t)BB * NN;
    float* scratch  = base + off; off += (size_t)BB * NN;
    int*   top_idx  = (int*)(base + off); off += 128;                // 80 used
    float* selected = base + off; off += (size_t)BB * KL * CC;
    float* sim_qn   = base + off; off += (size_t)BB * NN * KL;
    float* sim_km   = base + off; off += (size_t)BB * NN * KL;
    float* t_part   = base + off; off += (size_t)BB * 64 * KL * CC;
    float* t        = base + off; off += (size_t)BB * KL * CC;
    float* t2       = base + off; off += (size_t)BB * KL * CC;

    // 1) Q/K/V projections (WMMA bf16)
    proj_qkv_kernel<<<1536, 256, 0, stream>>>(query, Wq, bq, Wk, bk, Wv, bv,
                                              q, kbuf, vbuf);
    // 2) column sum of q  ->  qsum
    qsum_partial_kernel<<<BB * 32, 256, 0, stream>>>(q, qpart);
    qsum_reduce_kernel<<<BB, 256, 0, stream>>>(qpart, qsum);
    // 3) landmark scores
    score_kernel<<<(BB * NN) / 8, 256, 0, stream>>>(kbuf, qsum, score);
    // 4) top-40 + gather
    topk_gather_kernel<<<BB, 256, 0, stream>>>(score, kbuf, scratch, top_idx,
                                               selected);
    // 5) softmaxed landmark attention for q and k
    logits_softmax_kernel<<<(BB * NN) / 8, 256, 0, stream>>>(q, selected, sim_qn);
    logits_softmax_kernel<<<(BB * NN) / 8, 256, 0, stream>>>(kbuf, selected, sim_km);
    // 6) t = sim_km^T @ v  (split-K, deterministic)
    t_partial_kernel<<<BB * 64, 256, 0, stream>>>(sim_km, vbuf, t_part);
    t_reduce_kernel<<<(BB * KL * CC) / 256, 256, 0, stream>>>(t_part, t);
    // 7) fold output projection:  t2 = t @ Wo^T
    t2_kernel<<<BB * KL, 256, 0, stream>>>(t, Wo, t2);
    // 8) out = query + bo + sim_qn @ t2
    out_kernel<<<BB * (NN / 32), 256, 0, stream>>>(query, sim_qn, t2, bo, out);
}
